// HAN_66700842107204
// MI455X (gfx1250) — matrix operand order repaired and checked by reading, not compiled, not run on hardware
//
#include <hip/hip_runtime.h>
#include <hip/hip_bf16.h>
#include <math.h>

// ---------------------------------------------------------------------------
// HANConv (DBLP-style) on MI455X / gfx1250.
// Dense GEMMs -> v_wmma_f32_16x16x32_bf16 with fragment-order LDS staging so
// every operand fragment is a contiguous 32B ds_load_b128 pair.
// Edge attention -> atomic segment softmax + coalesced scatter (HBM-bound).
// Semantic ("group") attention -> tanh fused into GEMM epilogue (column sums
// only; never materializes the [M,N,C] tanh tensor).
// ---------------------------------------------------------------------------

typedef float  v8f    __attribute__((ext_vector_type(8)));
typedef __bf16 v16bf  __attribute__((ext_vector_type(16)));
typedef __bf16 bf16x4 __attribute__((ext_vector_type(4)));

#define BM 64
#define BN 64
#define KT 32
#define LDP (KT + 16)   // padded LDS row stride (48 bf16 = 96 B, 32B-aligned frags)
#define CH 256
#define NHEAD 8
#define HDIM 32

// ---------------------------------------------------------------------------
// W (fp32, row-major [K,256]) -> Wt (bf16, column-major [256][K])
// ---------------------------------------------------------------------------
__global__ void han_w_transpose_bf16(const float* __restrict__ W,
                                     __bf16* __restrict__ Wt, int K)
{
    int i = blockIdx.x * blockDim.x + threadIdx.x;   // over K*256, coalesced read
    if (i >= K * CH) return;
    int r = i >> 8, c = i & 255;
    Wt[(size_t)c * K + r] = (__bf16)W[i];
}

// ---------------------------------------------------------------------------
// Tiled GEMM: Cout[M,256] = act(A[M,K]) @ W[K,256] (+bias)
//   mode 0: write acc + bias to Cout
//   mode 1: s = tanh(acc + bias); atomicAdd per-column sums into colsum[256]
//   relu_in: apply relu to A elements on load (metapath outputs)
// Block: 128 threads (4 waves). Block tile 64x64, K-step 32.
// A staged into LDS in WMMA fragment element order; B staged column-major
// from the pre-transposed bf16 weights. All fragment reads: 32B contiguous.
// ---------------------------------------------------------------------------
__global__ __launch_bounds__(128)
void han_gemm_wmma(const float* __restrict__ A, const __bf16* __restrict__ Wt,
                   const float* __restrict__ bias, float* __restrict__ Cout,
                   float* __restrict__ colsum, int M, int K,
                   int relu_in, int mode)
{
    __shared__ __align__(16) __bf16 As[BM][LDP];   // 6 KB, rows in natural k order
    __shared__ __align__(16) __bf16 Bs[BN][LDP];   // 6 KB, column-major (k contig)

    const int tid  = threadIdx.x;
    const int wave = tid >> 5;           // 0..3 -> 16-row strip
    const int lane = tid & 31;
    const int row0 = blockIdx.x * BM;
    const int col0 = blockIdx.y * BN;    // 0,64,128,192

    v8f acc[4] = {};                     // 4 n-tiles of 16x16 f32

    const int mfrag = lane & 15;
    const int khalf = lane >> 4;

    for (int kb = 0; kb < K; kb += KT) {
        // ---- stage A tile: 64x32 fp32 -> bf16, fragment element order ----
        // fragment order: contiguous k (pos = k)
        #pragma unroll
        for (int ii = 0; ii < 4; ++ii) {
            int i = tid + ii * 128;          // 0..511
            int r = i >> 3;
            int c = (i & 7) * 4;
            int pos = c;
            int gr = row0 + r;
            float4 v = make_float4(0.f, 0.f, 0.f, 0.f);
            if (gr < M) v = *(const float4*)(A + (size_t)gr * K + kb + c);
            if (relu_in) {
                v.x = fmaxf(v.x, 0.f); v.y = fmaxf(v.y, 0.f);
                v.z = fmaxf(v.z, 0.f); v.w = fmaxf(v.w, 0.f);
            }
            bf16x4 b4 = { (__bf16)v.x, (__bf16)v.y, (__bf16)v.z, (__bf16)v.w };
            *(bf16x4*)(&As[r][pos]) = b4;
        }
        // ---- stage B tile: 64 cols x 32 k, bf16 from column-major Wt ----
        #pragma unroll
        for (int ii = 0; ii < 4; ++ii) {
            int i = tid + ii * 128;          // 0..511
            int cc = i >> 3;                 // 0..63
            int r4 = (i & 7) * 4;
            bf16x4 b4 = *(const bf16x4*)(Wt + (size_t)(col0 + cc) * K + kb + r4);
            *(bf16x4*)(&Bs[cc][r4]) = b4;
        }
        __syncthreads();

        // contiguous 32B fragment reads (ds_load_b128 x2 each)
        v16bf af = *(const v16bf*)(&As[wave * 16 + mfrag][khalf * 16]);
        #pragma unroll
        for (int t = 0; t < 4; ++t) {
            v16bf bfr = *(const v16bf*)(&Bs[t * 16 + mfrag][khalf * 16]);
            acc[t] = __builtin_amdgcn_wmma_f32_16x16x32_bf16(
                false, af, false, bfr, (short)0, acc[t], false, false);
        }
        __syncthreads();
    }

    // epilogue; D layout: VGPR r -> M = r + 8*(lane>=16), N = lane&15
    const int nlo = lane & 15;
    const int mhi = khalf * 8;
    if (mode == 0) {
        #pragma unroll
        for (int t = 0; t < 4; ++t) {
            int gcol = col0 + t * 16 + nlo;
            float b = bias[gcol];
            #pragma unroll
            for (int r = 0; r < 8; ++r) {
                int gm = row0 + wave * 16 + mhi + r;
                if (gm < M) Cout[(size_t)gm * CH + gcol] = acc[t][r] + b;
            }
        }
    } else {
        #pragma unroll
        for (int t = 0; t < 4; ++t) {
            int gcol = col0 + t * 16 + nlo;
            float b = bias[gcol];
            float s = 0.0f;
            #pragma unroll
            for (int r = 0; r < 8; ++r) {
                int gm = row0 + wave * 16 + mhi + r;
                if (gm < M) s += tanhf(acc[t][r] + b);
            }
            atomicAdd(&colsum[gcol], s);
        }
    }
}

// ---------------------------------------------------------------------------
// Utility fill
// ---------------------------------------------------------------------------
__global__ void han_fill(float* __restrict__ p, float v, long long n)
{
    long long i = blockIdx.x * (long long)blockDim.x + threadIdx.x;
    if (i < n) p[i] = v;
}

// ---------------------------------------------------------------------------
// alpha[n,h] = sum_d h[n, h*32+d] * att[h,d]
// ---------------------------------------------------------------------------
__global__ void han_node_alpha(const float* __restrict__ Hm,
                               const float* __restrict__ att,
                               float* __restrict__ out, int N)
{
    int i = blockIdx.x * blockDim.x + threadIdx.x;
    if (i >= N * NHEAD) return;
    int n = i >> 3, h = i & 7;
    const float* hv = Hm + (size_t)n * CH + h * HDIM;
    const float* av = att + h * HDIM;
    float s = 0.0f;
    #pragma unroll
    for (int d = 0; d < HDIM; d += 4) {
        float4 x = *(const float4*)(hv + d);
        float4 a = *(const float4*)(av + d);
        s += x.x * a.x + x.y * a.y + x.z * a.z + x.w * a.w;
    }
    out[i] = s;
}

__device__ __forceinline__ void atomicMaxFloat(float* addr, float v)
{
    if (v >= 0.0f) atomicMax((int*)addr, __float_as_int(v));
    else           atomicMin((unsigned int*)addr, (unsigned int)__float_as_int(v));
}

// pass 1: a = leaky_relu(asrc[src] + adst[dst]); store; atomic max per (dst,h)
__global__ void han_edge_alpha_max(const int* __restrict__ ei, int E_,
                                   const float* __restrict__ asrc,
                                   const float* __restrict__ adst,
                                   float* __restrict__ ebuf,
                                   float* __restrict__ amax)
{
    int i = blockIdx.x * blockDim.x + threadIdx.x;
    if (i >= E_ * NHEAD) return;
    int e = i >> 3, h = i & 7;
    int s = ei[e], d = ei[E_ + e];
    float a = asrc[s * NHEAD + h] + adst[d * NHEAD + h];
    a = (a > 0.0f) ? a : 0.2f * a;
    ebuf[i] = a;
    atomicMaxFloat(&amax[d * NHEAD + h], a);
}

// pass 2: e = exp(a - max); store; atomic sum per (dst,h)
__global__ void han_edge_exp_sum(const int* __restrict__ ei, int E_,
                                 const float* __restrict__ amax,
                                 float* __restrict__ ebuf,
                                 float* __restrict__ denom)
{
    int i = blockIdx.x * blockDim.x + threadIdx.x;
    if (i >= E_ * NHEAD) return;
    int e = i >> 3, h = i & 7;
    int d = ei[E_ + e];
    float ex = __expf(ebuf[i] - amax[d * NHEAD + h]);
    ebuf[i] = ex;
    atomicAdd(&denom[d * NHEAD + h], ex);
}

// pass 3: out[dst, c] += h_src[src, c] * e/(denom+eps)   (one channel per lane)
__global__ void han_edge_scatter(const int* __restrict__ ei, int E_,
                                 const float* __restrict__ Hsrc,
                                 const float* __restrict__ ebuf,
                                 const float* __restrict__ denom,
                                 float* __restrict__ out)
{
    long long i = blockIdx.x * (long long)blockDim.x + threadIdx.x;
    if (i >= (long long)E_ * CH) return;
    int e = (int)(i >> 8);
    int c = (int)(i & 255);
    int h = c >> 5;
    int s = ei[e], d = ei[E_ + e];
    float w = ebuf[e * NHEAD + h] / (denom[d * NHEAD + h] + 1e-16f);
    atomicAdd(&out[(size_t)d * CH + c], Hsrc[(size_t)s * CH + c] * w);
}

// semantic score: s_m = sum_c (colsum[m,c]/N) * q[c]; softmax over m in {0,1}
__global__ void han_group_score(const float* __restrict__ colsum,
                                const float* __restrict__ q,
                                float invN, float* __restrict__ attn)
{
    __shared__ float red0[CH], red1[CH];
    int c = threadIdx.x;
    red0[c] = colsum[c] * invN * q[c];
    red1[c] = colsum[CH + c] * invN * q[c];
    __syncthreads();
    for (int st = 128; st > 0; st >>= 1) {
        if (c < st) { red0[c] += red0[c + st]; red1[c] += red1[c + st]; }
        __syncthreads();
    }
    if (c == 0) {
        float s0 = red0[0], s1 = red1[0];
        float mx = fmaxf(s0, s1);
        float e0 = __expf(s0 - mx), e1 = __expf(s1 - mx);
        float inv = 1.0f / (e0 + e1);
        attn[0] = e0 * inv;
        attn[1] = e1 * inv;
    }
}

// out = attn0*relu(X0) + attn1*relu(X1)
__global__ void han_combine(const float* __restrict__ X0,
                            const float* __restrict__ X1,
                            const float* __restrict__ attn,
                            float* __restrict__ out, long long total)
{
    long long i = blockIdx.x * (long long)blockDim.x + threadIdx.x;
    if (i >= total) return;
    out[i] = attn[0] * fmaxf(X0[i], 0.0f) + attn[1] * fmaxf(X1[i], 0.0f);
}

// ---------------------------------------------------------------------------
// Host orchestration
// ---------------------------------------------------------------------------
extern "C" void kernel_launch(void* const* d_in, const int* in_sizes, int n_in,
                              void* d_out, int out_size, void* d_ws, size_t ws_size,
                              hipStream_t stream)
{
    const int NA = 20000, NP = 40000, E = 100000;
    const int C = CH, H = NHEAD, L = 3, PPL = 15;  // params per layer

    const float* x_author = (const float*)d_in[0];
    const float* x_paper  = (const float*)d_in[1];
    // per-layer param keys (dict insertion order):
    // 0 proj_author_w, 1 proj_author_b, 2 proj_paper_w, 3 proj_paper_b,
    // 4 k_w, 5 k_b, 6 q,
    // 7 att_src_ap, 8 att_dst_ap, 9 att_src_pa, 10 att_dst_pa,
    // 11 att_src_pp, 12 att_dst_pp, 13 att_src_aa, 14 att_dst_aa
    auto P = [&](int l, int k) { return (const float*)d_in[2 + l * PPL + k]; };
    const int* ei_ap = (const int*)d_in[2 + L * PPL + 0];
    const int* ei_pa = (const int*)d_in[2 + L * PPL + 1];
    const int* ei_pp = (const int*)d_in[2 + L * PPL + 2];
    const int* ei_aa = (const int*)d_in[2 + L * PPL + 3];

    // workspace carve-up (floats)
    float* w = (float*)d_ws;
    float* ha     = w; w += (size_t)NA * C;
    float* hp     = w; w += (size_t)NP * C;
    float* out_a0 = w; w += (size_t)NA * C;
    float* out_a1 = w; w += (size_t)NA * C;
    float* out_p0 = w; w += (size_t)NP * C;
    float* out_p1 = w; w += (size_t)NP * C;
    float* xa_buf = w; w += (size_t)NA * C;
    float* xp_buf = w; w += (size_t)NP * C;
    float* asrc   = w; w += (size_t)NP * H;
    float* adst   = w; w += (size_t)NP * H;
    float* amax   = w; w += (size_t)NP * H;
    float* denom  = w; w += (size_t)NP * H;
    float* ebuf   = w; w += (size_t)E * H;
    float* colsum = w; w += 2 * C;
    float* attn   = w; w += 4;
    __bf16* wt    = (__bf16*)w; w += (C * C) / 2;   // 256x256 bf16 transposed weights

    float* xa_final = (float*)d_out;
    float* xp_final = (float*)d_out + (size_t)NA * C;

    const float* xa_cur = x_author; int Ka = 128;
    const float* xp_cur = x_paper;  int Kp = 256;

    const int TB = 256;
    auto blocks = [&](long long n) { return (unsigned)((n + TB - 1) / TB); };

    for (int l = 0; l < L; ++l) {
        dim3 gA((NA + BM - 1) / BM, C / BN);
        dim3 gP((NP + BM - 1) / BM, C / BN);

        // node projections  ha = xa @ Wa + ba ; hp = xp @ Wp + bp
        han_w_transpose_bf16<<<blocks((long long)Ka * C), TB, 0, stream>>>(P(l, 0), wt, Ka);
        han_gemm_wmma<<<gA, 128, 0, stream>>>(xa_cur, wt, P(l, 1), ha, nullptr, NA, Ka, 0, 0);
        han_w_transpose_bf16<<<blocks((long long)Kp * C), TB, 0, stream>>>(P(l, 2), wt, Kp);
        han_gemm_wmma<<<gP, 128, 0, stream>>>(xp_cur, wt, P(l, 3), hp, nullptr, NP, Kp, 0, 0);

        struct ET { const int* ei; const float* Hs; const float* Hd;
                    int as, ad; int Ns, Nd; float* out; };
        const ET ets[4] = {
            { ei_ap, ha, hp,  7,  8, NA, NP, out_p0 },   // author -> paper
            { ei_pp, hp, hp, 11, 12, NP, NP, out_p1 },   // paper  -> paper
            { ei_pa, hp, ha,  9, 10, NP, NA, out_a0 },   // paper  -> author
            { ei_aa, ha, ha, 13, 14, NA, NA, out_a1 },   // author -> author
        };
        for (int t = 0; t < 4; ++t) {
            const ET& et = ets[t];
            han_fill<<<blocks((long long)et.Nd * H), TB, 0, stream>>>(amax, -INFINITY, (long long)et.Nd * H);
            han_fill<<<blocks((long long)et.Nd * H), TB, 0, stream>>>(denom, 0.0f, (long long)et.Nd * H);
            han_fill<<<blocks((long long)et.Nd * C), TB, 0, stream>>>(et.out, 0.0f, (long long)et.Nd * C);
            han_node_alpha<<<blocks((long long)et.Ns * H), TB, 0, stream>>>(et.Hs, P(l, et.as), asrc, et.Ns);
            han_node_alpha<<<blocks((long long)et.Nd * H), TB, 0, stream>>>(et.Hd, P(l, et.ad), adst, et.Nd);
            han_edge_alpha_max<<<blocks((long long)E * H), TB, 0, stream>>>(et.ei, E, asrc, adst, ebuf, amax);
            han_edge_exp_sum<<<blocks((long long)E * H), TB, 0, stream>>>(et.ei, E, amax, ebuf, denom);
            han_edge_scatter<<<blocks((long long)E * C), TB, 0, stream>>>(et.ei, E, et.Hs, ebuf, denom, et.out);
        }

        const bool last = (l == L - 1);

        // shared semantic weights for both node types this layer
        han_w_transpose_bf16<<<blocks((long long)C * C), TB, 0, stream>>>(P(l, 4), wt, C);

        // --- semantic group: authors ---
        han_fill<<<blocks(2 * C), TB, 0, stream>>>(colsum, 0.0f, 2 * C);
        han_gemm_wmma<<<gA, 128, 0, stream>>>(out_a0, wt, P(l, 5), nullptr, colsum,     NA, C, 1, 1);
        han_gemm_wmma<<<gA, 128, 0, stream>>>(out_a1, wt, P(l, 5), nullptr, colsum + C, NA, C, 1, 1);
        han_group_score<<<1, CH, 0, stream>>>(colsum, P(l, 6), 1.0f / NA, attn);
        float* xa_next = last ? xa_final : xa_buf;
        han_combine<<<blocks((long long)NA * C), TB, 0, stream>>>(out_a0, out_a1, attn, xa_next, (long long)NA * C);

        // --- semantic group: papers ---
        han_fill<<<blocks(2 * C), TB, 0, stream>>>(colsum, 0.0f, 2 * C);
        han_gemm_wmma<<<gP, 128, 0, stream>>>(out_p0, wt, P(l, 5), nullptr, colsum,     NP, C, 1, 1);
        han_gemm_wmma<<<gP, 128, 0, stream>>>(out_p1, wt, P(l, 5), nullptr, colsum + C, NP, C, 1, 1);
        han_group_score<<<1, CH, 0, stream>>>(colsum, P(l, 6), 1.0f / NP, attn);
        float* xp_next = last ? xp_final : xp_buf;
        han_combine<<<blocks((long long)NP * C), TB, 0, stream>>>(out_p0, out_p1, attn, xp_next, (long long)NP * C);

        xa_cur = xa_buf; Ka = C;
        xp_cur = xp_buf; Kp = C;
    }
    (void)in_sizes; (void)n_in; (void)out_size; (void)ws_size;
}
